// DySnakeConv_78451872629171
// MI455X (gfx1250) — compile-verified, matrix operand order checked by reading, not run on hardware
//
#include <hip/hip_runtime.h>
#include <hip/hip_bf16.h>
#include <stdint.h>

// MI455X / gfx1250, wave32. Implicit-GEMM 3x3 conv via bf16 hi/lo split
// (3-term emulation of fp32) on V_WMMA_F32_16X16X32_BF16.
//
// Roofline: 77.3 GFLOP / ~537 MB HBM => 144 FLOP/B; bf16 WMMA (K=32/instr)
// keeps the kernel at the ~23 us memory floor, fp32 WMMA (K=4) would not.
// Block tile = 64 C_OUT x 4h x 32w (halo 6/4 = 1.5x input traffic), with a
// 512-thread / 16-wave block so each wave keeps the low-pressure 2-chain
// rolling-fragment structure that fits under 256 VGPRs (no s_set_vgpr_msb).

typedef __attribute__((ext_vector_type(16))) __bf16 v16bf;
typedef __attribute__((ext_vector_type(8)))  float  v8f;
typedef __attribute__((ext_vector_type(8)))  unsigned int v8u;

#define B_   16
#define CIN  64
#define COUT 64
#define H_   256
#define W_   256
#define WT   36   // padded LDS row stride (dwords): 4*WT = 144 = 16 mod 64 banks
#define ROWS 6    // 4 output h-rows + 2 halo

// round-to-nearest-even f32 -> bf16
__device__ __forceinline__ unsigned bf16_rne(float f) {
  unsigned u = __float_as_uint(f);
  return (u + 0x7FFFu + ((u >> 16) & 1u)) >> 16;
}

// ---------------------------------------------------------------------------
// Kernel 1: split weights into bf16 hi/lo, prepacked in WMMA A-fragment order.
// Fragment f = c*4 + mi (c = (kh*3+kw)*2 + ciHalf); 32 lanes x 16 bf16/lane.
// 16-bit A layout (ISA 7.12.2): lane L<16 -> M=L, K = {0..7, 16..23};
//                               lane L>=16 -> M=L-16, K = {8..15, 24..31}.
// ---------------------------------------------------------------------------
__global__ void prepack_weights(const float* __restrict__ w,
                                unsigned short* __restrict__ whi,
                                unsigned short* __restrict__ wlo) {
  int id = blockIdx.x * blockDim.x + threadIdx.x;
  if (id >= 18 * 4 * 32) return;
  int lane = id & 31;
  int mi   = (id >> 5) & 3;
  int c    = id >> 7;
  int kh = c / 6, kw = (c / 2) % 3, ch = c & 1;
  int co = mi * 16 + (lane & 15);
  int kb = (lane & 16) ? 8 : 0;

  unsigned hiw[8], low[8];
#pragma unroll
  for (int v = 0; v < 8; ++v) {
    unsigned hp[2], lp[2];
#pragma unroll
    for (int e = 0; e < 2; ++e) {
      int j  = v * 2 + e;
      int kk = kb + (j < 8 ? j : j + 8);
      int ci = ch * 32 + kk;
      float x = w[((co * CIN + ci) * 3 + kh) * 3 + kw];
      unsigned h16 = bf16_rne(x);
      unsigned l16 = bf16_rne(x - __uint_as_float(h16 << 16));
      hp[e] = h16; lp[e] = l16;
    }
    hiw[v] = (hp[1] << 16) | hp[0];
    low[v] = (lp[1] << 16) | lp[0];
  }
  uint4* dh = reinterpret_cast<uint4*>(whi + id * 16);
  uint4* dl = reinterpret_cast<uint4*>(wlo + id * 16);
  dh[0] = make_uint4(hiw[0], hiw[1], hiw[2], hiw[3]);
  dh[1] = make_uint4(hiw[4], hiw[5], hiw[6], hiw[7]);
  dl[0] = make_uint4(low[0], low[1], low[2], low[3]);
  dl[1] = make_uint4(low[4], low[5], low[6], low[7]);
}

// ---------------------------------------------------------------------------
// Kernel 2: main conv. Block = 512 threads = 16 waves (one WGP, 4/SIMD).
// Output tile per block: all 64 C_OUT x (4 h-rows x 32 w).
// Wave wv: mi = wv&3 (C_OUT 16-row), np (w 16-col half), hb (h-row pair);
// two 16x16 f32 accumulators per wave (rows hb, hb+1), bias in the init.
//
// LDS: two arrays (hi, lo), each [6 rows][32 ci-pairs][WT] of packed dwords
// {bf16(ci_odd)<<16 | bf16(ci_even)} -- exactly the 16-bit WMMA operand word,
// so a B fragment VGPR is ONE ds_load_b32 (no perms).
//
// Per (kw, ci-half) chunk: sched_barrier fences prefetch to ONE chunk; the
// chunk's 3 A fragment pairs are hoisted explicitly (~48 VGPRs in flight),
// B row-fragments roll through a cur/nxt pair (32 dwords live).
// ---------------------------------------------------------------------------
__global__ void __launch_bounds__(512)
conv3x3_bf16x3_wmma(const float* __restrict__ x,
                    const float* __restrict__ bias,
                    const unsigned short* __restrict__ whi,
                    const unsigned short* __restrict__ wlo,
                    float* __restrict__ out) {
  __shared__ unsigned hiT[ROWS * 32 * WT];   // 27.6 KB
  __shared__ unsigned loT[ROWS * 32 * WT];   // 27.6 KB

  const int wt = blockIdx.x;              // 0..7  (w tiles of 32)
  const int ht = blockIdx.y;              // 0..63 (h tiles of 4)
  const int n  = blockIdx.z;              // 0..15
  const int h0 = ht * 4, w0 = wt * 32;
  const int t  = threadIdx.x;

  // ---- stage + split input tile: rows h0-1..h0+4, cols w0-1..w0+32 ----
  for (int e = t; e < ROWS * 32 * 34; e += 512) {
    int wl = e % 34;
    int q  = e / 34;                      // rr*32 + pair
    int gh = h0 - 1 + (q >> 5);
    int gw = w0 - 1 + wl;
    int ci0 = (q & 31) * 2;
    float x0 = 0.f, x1 = 0.f;
    if ((unsigned)gh < H_ && (unsigned)gw < W_) {
      const float* p = &x[((n * CIN + ci0) * H_ + gh) * W_ + gw];
      x0 = p[0];
      x1 = p[H_ * W_];
    }
    unsigned h0b = bf16_rne(x0), h1b = bf16_rne(x1);
    unsigned l0b = bf16_rne(x0 - __uint_as_float(h0b << 16));
    unsigned l1b = bf16_rne(x1 - __uint_as_float(h1b << 16));
    hiT[q * WT + wl] = (h1b << 16) | h0b;
    loT[q * WT + wl] = (l1b << 16) | l0b;
  }
  __syncthreads();

  const int lane  = t & 31;
  const int wv    = t >> 5;               // 0..15
  const int mi    = wv & 3;               // C_OUT tile
  const int np    = (wv >> 2) & 1;        // w-half (0: cols 0-15, 1: 16-31)
  const int hb    = (wv >> 3) * 2;        // h-row pair base: 0 or 2
  const int nl    = lane & 15;
  const int pb    = (lane & 16) ? 4 : 0;  // K pair sub-block per lane half
  const int cosel = (lane & 16) ? 8 : 0;  // C/D: lanes>=16 hold M=r+8

  v8f acc0, acc1;
#pragma unroll
  for (int r = 0; r < 8; ++r) {
    float bv = bias[mi * 16 + r + cosel];
    acc0[r] = bv; acc1[r] = bv;
  }

  const v16bf* afragH = reinterpret_cast<const v16bf*>(whi);
  const v16bf* afragL = reinterpret_cast<const v16bf*>(wlo);

#pragma unroll
  for (int kw = 0; kw < 3; ++kw) {
    const int wl = np * 16 + nl + kw;     // 0..33
#pragma unroll
    for (int ch = 0; ch < 2; ++ch) {      // ci half: K chunk of 32
      // fence: cap the scheduler's prefetch window to this chunk
      __builtin_amdgcn_sched_barrier(0);

      // A fragments for all three kh of this chunk, issued up front
      v16bf ah[3], al[3];
#pragma unroll
      for (int kh = 0; kh < 3; ++kh) {
        const int c = (kh * 3 + kw) * 2 + ch;
        ah[kh] = afragH[(c * 4 + mi) * 32 + lane];
        al[kh] = afragL[(c * 4 + mi) * 32 + lane];
      }

      const unsigned base0 = (unsigned)((ch * 16 + pb) * WT + wl)
                           + (unsigned)(hb * 32 * WT);

      // fragment for input row rr = hb
      v8u curh, curl;
#pragma unroll
      for (int v = 0; v < 8; ++v) {
        int pp = (v < 4) ? v : v + 4;     // pairs {0-3,8-11} (+pb per half)
        curh[v] = hiT[base0 + pp * WT];
        curl[v] = loT[base0 + pp * WT];
      }

#pragma unroll
      for (int kh = 0; kh < 3; ++kh) {
        // build fragment for row rr = hb+kh+1 (overlaps WMMAs on cur)
        const unsigned basen = base0 + (unsigned)((kh + 1) * 32 * WT);
        v8u nxth, nxtl;
#pragma unroll
        for (int v = 0; v < 8; ++v) {
          int pp = (v < 4) ? v : v + 4;
          nxth[v] = hiT[basen + pp * WT];
          nxtl[v] = loT[basen + pp * WT];
        }

        v16bf b0h = __builtin_bit_cast(v16bf, curh);
        v16bf b0l = __builtin_bit_cast(v16bf, curl);
        v16bf b1h = __builtin_bit_cast(v16bf, nxth);
        v16bf b1l = __builtin_bit_cast(v16bf, nxtl);

        acc0 = __builtin_amdgcn_wmma_f32_16x16x32_bf16(false, ah[kh], false, b0h, (short)0, acc0, false, false);
        acc1 = __builtin_amdgcn_wmma_f32_16x16x32_bf16(false, ah[kh], false, b1h, (short)0, acc1, false, false);
        acc0 = __builtin_amdgcn_wmma_f32_16x16x32_bf16(false, ah[kh], false, b0l, (short)0, acc0, false, false);
        acc1 = __builtin_amdgcn_wmma_f32_16x16x32_bf16(false, ah[kh], false, b1l, (short)0, acc1, false, false);
        acc0 = __builtin_amdgcn_wmma_f32_16x16x32_bf16(false, al[kh], false, b0h, (short)0, acc0, false, false);
        acc1 = __builtin_amdgcn_wmma_f32_16x16x32_bf16(false, al[kh], false, b1h, (short)0, acc1, false, false);

        curh = nxth;                      // rotate (register renaming)
        curl = nxtl;
      }
    }
  }

  // ---- store: lanes 0-15 write 16 consecutive w per reg (64B segments) ----
  const int wg = w0 + np * 16 + nl;
#pragma unroll
  for (int tt = 0; tt < 2; ++tt) {
    const int hg = h0 + hb + tt;
    v8f acc = tt ? acc1 : acc0;
#pragma unroll
    for (int r = 0; r < 8; ++r) {
      const int co = mi * 16 + r + cosel;
      out[((n * COUT + co) * H_ + hg) * W_ + wg] = acc[r];
    }
  }
}

extern "C" void kernel_launch(void* const* d_in, const int* in_sizes, int n_in,
                              void* d_out, int out_size, void* d_ws, size_t ws_size,
                              hipStream_t stream) {
  const float* x    = (const float*)d_in[0];
  const float* w    = (const float*)d_in[1];
  const float* bias = (const float*)d_in[2];
  float* out = (float*)d_out;

  // d_ws: [36864 x u16 hi][36864 x u16 lo] = 147,456 B of prepacked weights
  unsigned short* whi = (unsigned short*)d_ws;
  unsigned short* wlo = whi + 18 * 4 * 32 * 16;

  prepack_weights<<<9, 256, 0, stream>>>(w, whi, wlo);

  dim3 grid(W_ / 32, H_ / 4, B_);         // 8 x 64 x 16 blocks, 512 thr each
  conv3x3_bf16x3_wmma<<<grid, 512, 0, stream>>>(x, bias, whi, wlo, out);
}
